// GWAN_40261023432900
// MI455X (gfx1250) — matrix operand order repaired and checked by reading, not compiled, not run on hardware
//
#include <hip/hip_runtime.h>
#include <hip/hip_bf16.h>

#define N_NODES 1024
#define F_DIM   1024
#define H_DIM   512
#define E_EDGES 32768
#define B_GR    16
#define OUT_DIM 10
#define ZC      1536      /* F_DIM + H_DIM */
#define FC1_OUT 512

typedef __attribute__((ext_vector_type(16))) __bf16 v16bf;
typedef __attribute__((ext_vector_type(8)))  __bf16 v8bf;
typedef __attribute__((ext_vector_type(8)))  float  v8f;

__device__ __forceinline__ float sigmoidf(float x) { return 1.0f / (1.0f + __expf(-x)); }

// ---------------------------------------------------------------- stage 1:
// lifting split + attention gate; writes h and initializes agg = h
__global__ void k_front(const float* __restrict__ x, const float* __restrict__ watt,
                        float* __restrict__ h, float* __restrict__ agg) {
    int idx = blockIdx.x * blockDim.x + threadIdx.x;
    if (idx >= N_NODES * H_DIM) return;
    int n = idx / H_DIM, c = idx % H_DIM;
    float xe = x[n * F_DIM + 2 * c];
    float xo = x[n * F_DIM + 2 * c + 1];
    const float is2 = 0.70710678118654752440f;  // 1/sqrt(2)
    float low = (xe + xo) * is2, high = (xe - xo) * is2;
    float s = sigmoidf(low * watt[0] + high * watt[1]);
    float hv = s * low + (1.0f - s) * high;
    h[idx] = hv;
    agg[idx] = hv;
}

// ---------------------------------------------------------------- stage 2:
// edge scatter-add: agg[dst] += h[src]   (134 MB total -> ~6us at 23.3 TB/s)
__global__ void k_scatter(const float* __restrict__ h, const int* __restrict__ ei,
                          float* __restrict__ agg) {
    int t = blockIdx.x * blockDim.x + threadIdx.x;
    if (t >= E_EDGES * (H_DIM / 4)) return;
    int e = t / (H_DIM / 4), q = t % (H_DIM / 4);
    int src = ei[e], dst = ei[E_EDGES + e];
    const float4 v = *(const float4*)(h + src * H_DIM + 4 * q);
    float* p = agg + dst * H_DIM + 4 * q;
    atomicAdd(p + 0, v.x); atomicAdd(p + 1, v.y);
    atomicAdd(p + 2, v.z); atomicAdd(p + 3, v.w);
}

// ---------------------------------------------------------------- prep:
// silu(agg)->bf16 (GEMM A), weights -> bf16 (GEMM B), 1/scale precompute
__global__ void k_prep(const float* __restrict__ agg, const float* __restrict__ wscale,
                       const float* __restrict__ wbase, const float* __restrict__ fc1w,
                       __bf16* __restrict__ siluA, __bf16* __restrict__ baseWbf,
                       float* __restrict__ rscale, __bf16* __restrict__ fc1wbf) {
    int idx = blockIdx.x * blockDim.x + threadIdx.x;
    if (idx < N_NODES * H_DIM) {
        float a = agg[idx];
        siluA[idx] = (__bf16)(a * sigmoidf(a));
    }
    if (idx < H_DIM * H_DIM) {
        rscale[idx] = 1.0f / wscale[idx];
        baseWbf[idx] = (__bf16)wbase[idx];
    }
    if (idx < FC1_OUT * ZC) {
        fc1wbf[idx] = (__bf16)fc1w[idx];
    }
}

// ---------------------------------------------------------------- stage 3:
// Mexican-hat wavelet contraction: the transcendental-bound hot loop.
// 8 nodes per block so each (o,i) weight triple is reused 8x from registers.
__global__ void __launch_bounds__(256) k_wavelet(const float* __restrict__ agg,
                                                 const float* __restrict__ trans,
                                                 const float* __restrict__ rscale,
                                                 const float* __restrict__ wavw,
                                                 float* __restrict__ wavbase) {
    __shared__ float arow[8][H_DIM];
    int nb = blockIdx.x >> 1;   // node group: 0..127
    int og = blockIdx.x & 1;    // output half: 0..1
    int tid = threadIdx.x;
    int n0 = nb * 8;
    for (int i = tid; i < 8 * H_DIM; i += 256)
        arow[i >> 9][i & 511] = agg[(n0 + (i >> 9)) * H_DIM + (i & 511)];
    __syncthreads();

    int o = og * 256 + tid;
    const float* tp = trans + o * H_DIM;
    const float* rp = rscale + o * H_DIM;
    const float* wp = wavw + o * H_DIM;
    float acc[8] = {0.f, 0.f, 0.f, 0.f, 0.f, 0.f, 0.f, 0.f};
    for (int i = 0; i < H_DIM; ++i) {
        float t = tp[i], rs = rp[i], w = wp[i];
#pragma unroll
        for (int j = 0; j < 8; ++j) {
            float xs = (arow[j][i] - t) * rs;   // LDS read is a broadcast (same addr)
            float x2 = xs * xs;
            acc[j] += (1.0f - x2) * __expf(-0.5f * x2) * w;   // v_exp_f32
        }
    }
    const float MHC = 0.8673250705840776f;
#pragma unroll
    for (int j = 0; j < 8; ++j)
        wavbase[(n0 + j) * H_DIM + o] = MHC * acc[j];
}

// ---------------------------------------------------------------- stage 4:
// base = silu(agg) @ Wb^T via v_wmma_f32_16x16x32_bf16, accumulated into wavbase.
// One wave per 16x16 tile; A/B fragments both use contiguous k = 16*(lane/16) + s.
__global__ void __launch_bounds__(128) k_base_gemm(const __bf16* __restrict__ A,
                                                   const __bf16* __restrict__ Bm,
                                                   float* __restrict__ C) {
    int wave = (blockIdx.x * blockDim.x + threadIdx.x) >> 5;
    int lane = threadIdx.x & 31;
    int tileRow = wave >> 5;          // 64 row tiles
    int tileCol = wave & 31;          // 32 col tiles
    int kh = lane >> 4;
    int r = tileRow * 16 + (lane & 15);
    int col = tileCol * 16 + (lane & 15);
    v8f acc = {0.f, 0.f, 0.f, 0.f, 0.f, 0.f, 0.f, 0.f};
    for (int kt = 0; kt < H_DIM / 32; ++kt) {
        int k0 = kt * 32;
        const __bf16* pA = A + r * H_DIM + k0 + kh * 16;
        v8bf a0 = *(const v8bf*)(pA);
        v8bf a1 = *(const v8bf*)(pA + 8);
        v16bf av = __builtin_shufflevector(a0, a1, 0, 1, 2, 3, 4, 5, 6, 7,
                                           8, 9, 10, 11, 12, 13, 14, 15);
        const __bf16* pB = Bm + col * H_DIM + k0 + kh * 16;
        v16bf bv = *(const v16bf*)(pB);
        __builtin_prefetch(pA + 32, 0, 1);          // global_prefetch_b8
        __builtin_prefetch(pB + 32, 0, 1);
        acc = __builtin_amdgcn_wmma_f32_16x16x32_bf16(false, av, false, bv,
                                                      (short)0, acc, false, false);
    }
#pragma unroll
    for (int v = 0; v < 8; ++v) {
        int row = tileRow * 16 + v + 8 * kh;
        int cc = tileCol * 16 + (lane & 15);
        C[row * H_DIM + cc] += acc[v];
    }
}

// ---------------------------------------------------------------- stage 5:
// column stats. mode 1: triple-BN collapsed analytically (bn(bn(.)) keeps zero
// mean so conv path reduces to (y-m)*g). mode 0: plain BN for x columns.
__global__ void k_colstats(const float* __restrict__ src, int ncols, int mode,
                           float* __restrict__ mout, float* __restrict__ gout) {
    __shared__ float ss[256], sq[256];
    int c = blockIdx.x, tid = threadIdx.x;
    float s = 0.f, q = 0.f;
    for (int n = tid; n < N_NODES; n += 256) {
        float v = src[n * ncols + c];
        s += v; q += v * v;
    }
    ss[tid] = s; sq[tid] = q; __syncthreads();
    for (int w = 128; w > 0; w >>= 1) {
        if (tid < w) { ss[tid] += ss[tid + w]; sq[tid] += sq[tid + w]; }
        __syncthreads();
    }
    if (tid == 0) {
        const float eps = 1e-5f;
        float mean = ss[0] * (1.0f / N_NODES);
        float var = sq[0] * (1.0f / N_NODES) - mean * mean;
        float g;
        if (mode == 0) {
            g = rsqrtf(var + eps);
        } else {
            float r1 = rsqrtf(var + eps);
            float f = r1 * rsqrtf(var * r1 * r1 + eps);   // bn2 of zero-mean bn1
            g = f * rsqrtf(var * f * f + eps);            // bn3 of zero-mean conv
        }
        mout[c] = mean; gout[c] = g;
    }
}

__global__ void k_zero(float* __restrict__ pooled, float* __restrict__ counts) {
    int idx = blockIdx.x * blockDim.x + threadIdx.x;
    if (idx < B_GR * ZC) pooled[idx] = 0.f;
    if (idx < B_GR) counts[idx] = 0.f;
}

__global__ void k_count(const int* __restrict__ batch, float* __restrict__ counts) {
    int n = blockIdx.x * blockDim.x + threadIdx.x;
    if (n < N_NODES) atomicAdd(&counts[batch[n]], 1.0f);
}

// normalize-on-the-fly + segment sum (z never materialized)
__global__ void k_pool(const float* __restrict__ x, const float* __restrict__ wb,
                       const int* __restrict__ batch,
                       const float* __restrict__ mx, const float* __restrict__ gx,
                       const float* __restrict__ m1, const float* __restrict__ g,
                       float* __restrict__ pooled) {
    int idx = blockIdx.x * blockDim.x + threadIdx.x;
    if (idx >= N_NODES * ZC) return;
    int n = idx / ZC, c = idx % ZC;
    float val;
    if (c < F_DIM) {
        val = (x[n * F_DIM + c] - mx[c]) * gx[c];
    } else {
        int cc = c - F_DIM;
        val = (wb[n * H_DIM + cc] - m1[cc]) * g[cc];
    }
    atomicAdd(&pooled[batch[n] * ZC + c], val);
}

__global__ void k_poolfin(float* __restrict__ pooled, const float* __restrict__ counts,
                          __bf16* __restrict__ pooledbf) {
    int idx = blockIdx.x * blockDim.x + threadIdx.x;
    if (idx >= B_GR * ZC) return;
    int b = idx / ZC;
    float p = pooled[idx] / fmaxf(counts[b], 1.0f);
    pooled[idx] = p;
    pooledbf[idx] = (__bf16)p;
}

// ---------------------------------------------------------------- stage 6:
// fc1 = relu(pooled @ W1^T + b1): M=16 exactly one WMMA tile row; 32 col tiles.
__global__ void __launch_bounds__(256) k_fc1(const __bf16* __restrict__ A,
                                             const __bf16* __restrict__ Bm,
                                             const float* __restrict__ bias,
                                             float* __restrict__ h1) {
    int wave = threadIdx.x >> 5;               // 0..7
    int lane = threadIdx.x & 31;
    int tileCol = blockIdx.x * 8 + wave;       // 0..31
    int kh = lane >> 4;
    int rA = lane & 15;
    int col = tileCol * 16 + (lane & 15);
    v8f acc = {0.f, 0.f, 0.f, 0.f, 0.f, 0.f, 0.f, 0.f};
    for (int kt = 0; kt < ZC / 32; ++kt) {
        int k0 = kt * 32;
        const __bf16* pA = A + rA * ZC + k0 + kh * 16;
        v8bf a0 = *(const v8bf*)(pA);
        v8bf a1 = *(const v8bf*)(pA + 8);
        v16bf av = __builtin_shufflevector(a0, a1, 0, 1, 2, 3, 4, 5, 6, 7,
                                           8, 9, 10, 11, 12, 13, 14, 15);
        const __bf16* pB = Bm + col * ZC + k0 + kh * 16;
        v16bf bv = *(const v16bf*)(pB);
        __builtin_prefetch(pB + 32, 0, 1);
        acc = __builtin_amdgcn_wmma_f32_16x16x32_bf16(false, av, false, bv,
                                                      (short)0, acc, false, false);
    }
    float b = bias[col];
#pragma unroll
    for (int v = 0; v < 8; ++v) {
        int row = v + 8 * kh;                  // all 16 rows valid (B_GR == 16)
        h1[row * FC1_OUT + col] = fmaxf(acc[v] + b, 0.0f);
    }
}

// fc2: 16x10x512, trivial
__global__ void k_fc2(const float* __restrict__ h1, const float* __restrict__ w,
                      const float* __restrict__ b, float* __restrict__ out) {
    int t = blockIdx.x * blockDim.x + threadIdx.x;
    if (t >= B_GR * OUT_DIM) return;
    int bg = t / OUT_DIM, o = t % OUT_DIM;
    float acc = b[o];
    for (int k = 0; k < FC1_OUT; ++k)
        acc += h1[bg * FC1_OUT + k] * w[o * FC1_OUT + k];
    out[bg * OUT_DIM + o] = acc;
}

// ----------------------------------------------------------------
extern "C" void kernel_launch(void* const* d_in, const int* in_sizes, int n_in,
                              void* d_out, int out_size, void* d_ws, size_t ws_size,
                              hipStream_t stream) {
    const float* x      = (const float*)d_in[0];
    const float* w_att  = (const float*)d_in[1];
    const float* wscale = (const float*)d_in[2];
    const float* wtrans = (const float*)d_in[3];
    const float* wwav   = (const float*)d_in[4];
    const float* wbase  = (const float*)d_in[5];
    const float* fc1w   = (const float*)d_in[6];
    const float* fc1b   = (const float*)d_in[7];
    const float* fc2w   = (const float*)d_in[8];
    const float* fc2b   = (const float*)d_in[9];
    const int*   eidx   = (const int*)d_in[10];
    const int*   batch  = (const int*)d_in[11];
    float* out = (float*)d_out;

    // workspace carve-up (256B aligned)
    char* ws = (char*)d_ws;
    size_t off = 0;
    auto carve = [&](size_t bytes) -> char* {
        off = (off + 255) & ~(size_t)255;
        char* p = ws + off; off += bytes; return p;
    };
    float*  hbuf     = (float*)carve(N_NODES * H_DIM * 4);
    float*  agg      = (float*)carve(N_NODES * H_DIM * 4);
    __bf16* siluA    = (__bf16*)carve(N_NODES * H_DIM * 2);
    __bf16* baseWbf  = (__bf16*)carve(H_DIM * H_DIM * 2);
    float*  rscale   = (float*)carve(H_DIM * H_DIM * 4);
    float*  wavbase  = (float*)carve(N_NODES * H_DIM * 4);
    float*  m1       = (float*)carve(H_DIM * 4);
    float*  gcv      = (float*)carve(H_DIM * 4);
    float*  mx       = (float*)carve(F_DIM * 4);
    float*  gx       = (float*)carve(F_DIM * 4);
    float*  pooled   = (float*)carve(B_GR * ZC * 4);
    float*  counts   = (float*)carve(B_GR * 4);
    __bf16* pooledbf = (__bf16*)carve(B_GR * ZC * 2);
    __bf16* fc1wbf   = (__bf16*)carve(FC1_OUT * ZC * 2);
    float*  h1       = (float*)carve(B_GR * FC1_OUT * 4);
    (void)ws_size; (void)in_sizes; (void)n_in; (void)out_size;

    // stage 1: lifting + gate
    k_front<<<(N_NODES * H_DIM + 255) / 256, 256, 0, stream>>>(x, w_att, hbuf, agg);
    // stage 2: edge scatter
    k_scatter<<<(E_EDGES * (H_DIM / 4) + 255) / 256, 256, 0, stream>>>(hbuf, eidx, agg);
    // prep: bf16 conversions + reciprocal scale
    k_prep<<<(FC1_OUT * ZC + 255) / 256, 256, 0, stream>>>(agg, wscale, wbase, fc1w,
                                                           siluA, baseWbf, rscale, fc1wbf);
    // stage 3: wavelet contraction (transcendental-bound hot loop)
    k_wavelet<<<(N_NODES / 8) * (H_DIM / 256), 256, 0, stream>>>(agg, wtrans, rscale,
                                                                 wwav, wavbase);
    // stage 4: base GEMM via WMMA, accumulated into wavbase
    k_base_gemm<<<(64 * 32) / 4, 128, 0, stream>>>(siluA, baseWbf, wavbase);
    // stage 5: fused triple-BN stats + pooling
    k_colstats<<<H_DIM, 256, 0, stream>>>(wavbase, H_DIM, 1, m1, gcv);
    k_colstats<<<F_DIM, 256, 0, stream>>>(x, F_DIM, 0, mx, gx);
    k_zero<<<(B_GR * ZC + 255) / 256, 256, 0, stream>>>(pooled, counts);
    k_count<<<(N_NODES + 255) / 256, 256, 0, stream>>>(batch, counts);
    k_pool<<<(N_NODES * ZC + 255) / 256, 256, 0, stream>>>(x, wavbase, batch,
                                                           mx, gx, m1, gcv, pooled);
    k_poolfin<<<(B_GR * ZC + 255) / 256, 256, 0, stream>>>(pooled, counts, pooledbf);
    // stage 6: fc1 (WMMA) + fc2
    k_fc1<<<4, 256, 0, stream>>>(pooledbf, fc1wbf, fc1b, h1);
    k_fc2<<<1, 256, 0, stream>>>(h1, fc2w, fc2b, out);
}
